// AttentionBlock_1013612282001
// MI455X (gfx1250) — matrix operand order repaired and checked by reading, not compiled, not run on hardware
//
#include <hip/hip_runtime.h>
#include <hip/hip_bf16.h>

// ---------------------------------------------------------------------------
// Types for CDNA5 WMMA (wave32): v_wmma_f32_16x16x32_bf16
// ---------------------------------------------------------------------------
typedef __bf16 bf16_t;
typedef bf16_t bf16x16 __attribute__((ext_vector_type(16)));
typedef bf16_t bf16x8  __attribute__((ext_vector_type(8)));
typedef float  f32x8   __attribute__((ext_vector_type(8)));

#define NUM_HEADS 16
#define EMB       1024
#define SEQ       2048
#define BATCH     4
#define DH        64
#define NQKV      (3 * EMB)

__device__ __forceinline__ f32x8 wmma_bf16(bf16x16 a, bf16x16 b, f32x8 c) {
    return __builtin_amdgcn_wmma_f32_16x16x32_bf16(
        false, a, false, b, (short)0, c, false, false);
}

// A-fragment (16x32 [m][k] tile row in LDS): lanes0-15 K=j / 16+j, lanes16-31 +8
__device__ __forceinline__ bf16x16 frag_a(const bf16_t* rowk, int hi) {
    bf16x16 a;
    ((bf16x8*)&a)[0] = *(const bf16x8*)(rowk + 8 * hi);
    ((bf16x8*)&a)[1] = *(const bf16x8*)(rowk + 16 + 8 * hi);
    return a;
}
// B-fragment from [n][k] tile row: lanes0-15 K=j / 16+j, lanes16-31 +8, N = lane&15
__device__ __forceinline__ bf16x16 frag_b(const bf16_t* rowk, int hi) {
    bf16x16 b;
    ((bf16x8*)&b)[0] = *(const bf16x8*)(rowk + 8 * hi);
    ((bf16x8*)&b)[1] = *(const bf16x8*)(rowk + 16 + 8 * hi);
    return b;
}

// CDNA5 async copy: global -> LDS, 16B per lane, tracked by ASYNCcnt.
template <typename T>
__device__ __forceinline__ unsigned lds_addr(T* p) {
    return (unsigned)(unsigned long long)p;   // low 32 bits of flat addr = LDS offset
}
__device__ __forceinline__ void async_b128(unsigned lds_off, const void* base,
                                           unsigned goff) {
    asm volatile("global_load_async_to_lds_b128 %0, %1, %2"
                 :: "v"(lds_off), "v"(goff), "s"(base) : "memory");
}
__device__ __forceinline__ void wait_async() {
    asm volatile("s_wait_asynccnt 0x0" ::: "memory");
}

// ---------------------------------------------------------------------------
// Kernel 0a: elementwise f32 -> bf16 (for x).  8 elems/thread.
// ---------------------------------------------------------------------------
__global__ __launch_bounds__(256)
void cvt_kernel(const float* __restrict__ in, bf16_t* __restrict__ out) {
    int i = (blockIdx.x * 256 + threadIdx.x) * 8;
    float4 f0 = *(const float4*)(in + i);
    float4 f1 = *(const float4*)(in + i + 4);
    bf16x8 v;
    v[0] = (bf16_t)f0.x; v[1] = (bf16_t)f0.y; v[2] = (bf16_t)f0.z; v[3] = (bf16_t)f0.w;
    v[4] = (bf16_t)f1.x; v[5] = (bf16_t)f1.y; v[6] = (bf16_t)f1.z; v[7] = (bf16_t)f1.w;
    *(bf16x8*)(out + i) = v;
}

// ---------------------------------------------------------------------------
// Kernel 0b: transpose + convert weights: out[n][k] = (bf16) in[k][n]
// ---------------------------------------------------------------------------
__global__ __launch_bounds__(256)
void cvt_tr_kernel(const float* __restrict__ in, bf16_t* __restrict__ out,
                   int K, int N) {
    __shared__ float t_s[32][33];
    const int k0 = blockIdx.y * 32, n0 = blockIdx.x * 32;
    const int tx = threadIdx.x & 31, ty = threadIdx.x >> 5;  // 32 x 8
    for (int i = 0; i < 4; ++i) {
        int k = ty + i * 8;
        t_s[k][tx] = in[(size_t)(k0 + k) * N + n0 + tx];
    }
    __syncthreads();
    for (int i = 0; i < 4; ++i) {
        int n = ty + i * 8;
        out[(size_t)(n0 + n) * K + k0 + tx] = (bf16_t)t_s[tx][n];
    }
}

// ===========================================================================
// GEMM core: 128x128 block tile, 8 waves x (32x64), explicit double buffer.
// KERNEL(body) is instantiated twice below with different epilogues.
// ===========================================================================
#define GEMM_BODY(A_PTR, B_PTR)                                                \
    __shared__ alignas(32) bf16_t a_s[2][128][32];                             \
    __shared__ alignas(32) bf16_t b_s[2][128][32];                             \
    const int tid = threadIdx.x;                                               \
    const int lane = tid & 31, wave = tid >> 5;                                \
    const int lo = lane & 15, hi = lane >> 4;                                  \
    const int m0 = blockIdx.y * 128, n0 = blockIdx.x * 128;                    \
    const int wm = wave >> 1, wn = wave & 1;                                   \
    const unsigned aB = lds_addr(&a_s[0][0][0]);                               \
    const unsigned bB = lds_addr(&b_s[0][0][0]);                               \
    f32x8 c00 = {}, c01 = {}, c02 = {}, c03 = {};                              \
    f32x8 c10 = {}, c11 = {}, c12 = {}, c13 = {};                              \
    auto stage = [&](unsigned boff, int k0) {                                  \
        _Pragma("unroll")                                                      \
        for (int it = 0; it < 2; ++it) {                                       \
            int tt = tid + it * 256;                                           \
            int r = tt >> 2, q = (tt & 3) * 16;                                \
            async_b128(aB + boff + r * 64 + q, A_PTR,                          \
                       (unsigned)((m0 + r) * EMB + k0) * 2 + q);               \
            async_b128(bB + boff + r * 64 + q, B_PTR,                          \
                       (unsigned)((n0 + r) * EMB + k0) * 2 + q);               \
        }                                                                      \
    };                                                                         \
    auto step = [&](const bf16_t (*as)[32], const bf16_t (*bs)[32],            \
                    int k_next, unsigned boff_next, bool do_stage) {           \
        wait_async();                                                          \
        __syncthreads();              /* current tile visible to all waves */  \
        if (do_stage) stage(boff_next, k_next);   /* prefetch overlaps WMMA */ \
        bf16x16 a0 = frag_a(&as[wm * 32 + lo][0], hi);                         \
        bf16x16 a1 = frag_a(&as[wm * 32 + 16 + lo][0], hi);                    \
        bf16x16 b0 = frag_b(&bs[wn * 64 + lo][0], hi);                         \
        bf16x16 b1 = frag_b(&bs[wn * 64 + 16 + lo][0], hi);                    \
        bf16x16 b2 = frag_b(&bs[wn * 64 + 32 + lo][0], hi);                    \
        bf16x16 b3 = frag_b(&bs[wn * 64 + 48 + lo][0], hi);                    \
        c00 = wmma_bf16(a0, b0, c00); c10 = wmma_bf16(a1, b0, c10);            \
        c01 = wmma_bf16(a0, b1, c01); c11 = wmma_bf16(a1, b1, c11);            \
        c02 = wmma_bf16(a0, b2, c02); c12 = wmma_bf16(a1, b2, c12);            \
        c03 = wmma_bf16(a0, b3, c03); c13 = wmma_bf16(a1, b3, c13);            \
        __syncthreads();              /* reads done before buf overwrite */    \
    };                                                                         \
    stage(0, 0);                                                               \
    for (int ks = 0; ks < EMB / 32; ks += 2) {                                 \
        step(a_s[0], b_s[0], (ks + 1) * 32, 8192, ks + 1 < EMB / 32);          \
        step(a_s[1], b_s[1], (ks + 2) * 32, 0,    ks + 2 < EMB / 32);          \
    }                                                                          \
    f32x8 accs[2][4] = {{c00, c01, c02, c03}, {c10, c11, c12, c13}};

// ---------------------------------------------------------------------------
// Kernel 1: QKV projection.  C[8192][3072] = xh @ wqkvT^T   (both bf16)
// Epilogue scatters Q (pre-scaled 1/8), K [bh][t][dh], V^T [bh][dh][t].
// ---------------------------------------------------------------------------
__global__ __launch_bounds__(256)
void qkv_kernel(const bf16_t* __restrict__ xh, const bf16_t* __restrict__ wT,
                bf16_t* __restrict__ qb, bf16_t* __restrict__ kb,
                bf16_t* __restrict__ vb) {
    GEMM_BODY(xh, wT)
    for (int mt = 0; mt < 2; ++mt)
        for (int nt = 0; nt < 4; ++nt) {
            f32x8 av = accs[mt][nt];
            int n_g = n0 + wn * 64 + nt * 16 + lo;
            int which = n_g >> 10;     // 0=Q 1=K 2=V
            int c = n_g & 1023;
            int h = c >> 6, dh = c & 63;
            for (int r = 0; r < 8; ++r) {
                int m_g = m0 + wm * 32 + mt * 16 + r + 8 * hi;
                int b = m_g >> 11, s = m_g & 2047;
                int bh = b * NUM_HEADS + h;
                float v = av[r];
                if (which == 0)
                    qb[((size_t)bh * SEQ + s) * DH + dh] = (bf16_t)(v * 0.125f);
                else if (which == 1)
                    kb[((size_t)bh * SEQ + s) * DH + dh] = (bf16_t)v;
                else
                    vb[((size_t)bh * DH + dh) * SEQ + s] = (bf16_t)v;
            }
        }
}

// ---------------------------------------------------------------------------
// Kernel 3: output projection.  out[8192][1024] = ob(bf16) @ woutT^T, f32 out.
// ---------------------------------------------------------------------------
__global__ __launch_bounds__(256)
void proj_kernel(const bf16_t* __restrict__ attn, const bf16_t* __restrict__ wT,
                 float* __restrict__ out) {
    GEMM_BODY(attn, wT)
    for (int mt = 0; mt < 2; ++mt)
        for (int nt = 0; nt < 4; ++nt) {
            f32x8 av = accs[mt][nt];
            int n_g = n0 + wn * 64 + nt * 16 + lo;
            for (int r = 0; r < 8; ++r) {
                int m_g = m0 + wm * 32 + mt * 16 + r + 8 * hi;
                out[(size_t)m_g * EMB + n_g] = av[r];
            }
        }
}

// ---------------------------------------------------------------------------
// Kernel 2: flash attention, transposed formulation.
// Block = 128 query rows of one (batch,head); wave w owns query cols
// m = w*16 + (lane&15).  S^T = K Q^T and O^T = V^T P^T: online softmax reduces
// in-register (one shfl_xor(16)); alpha / 1/l are one scalar per lane.
// K/V tiles double-buffered (explicit phases), async-staged; 2 barriers/tile.
// ---------------------------------------------------------------------------
__global__ __launch_bounds__(256)
void attn_kernel(const bf16_t* __restrict__ qb, const bf16_t* __restrict__ kb,
                 const bf16_t* __restrict__ vb, bf16_t* __restrict__ ob) {
    __shared__ alignas(32) bf16_t q_s[128][64];     // [m][dh]  (B for S^T)
    __shared__ alignas(32) bf16_t k_s[2][64][64];   // [t][dh]  (A for S^T)
    __shared__ alignas(32) bf16_t v_s[2][64][64];   // [dh][t]  (A for O^T)
    __shared__ alignas(32) bf16_t p_s[128][64];     // [m][t]   (B for O^T), wave-private rows

    const int tid = threadIdx.x;
    const int lane = tid & 31, wave = tid >> 5;
    const int lo = lane & 15, hi = lane >> 4;
    const int bh = blockIdx.y;
    const int s0 = blockIdx.x * 128;
    const int m_base = wave * 16;
    const unsigned qB = lds_addr(&q_s[0][0]);
    const unsigned kB = lds_addr(&k_s[0][0][0]);
    const unsigned vB = lds_addr(&v_s[0][0][0]);

    auto stage_kv = [&](int t1, unsigned boff) {
        #pragma unroll
        for (int it = 0; it < 2; ++it) {
            int tt = tid + it * 256;
            int r = tt >> 3, q = (tt & 7) * 16;
            async_b128(kB + boff + r * 128 + q, kb,
                       (unsigned)((bh * SEQ + t1 + r) * DH) * 2 + q);
            async_b128(vB + boff + r * 128 + q, vb,
                       (unsigned)((bh * DH + r) * SEQ + t1) * 2 + q);
        }
    };

    // stage Q tile (resident) + K/V tile 0
    #pragma unroll
    for (int it = 0; it < 4; ++it) {
        int tt = tid + it * 256;
        int r = tt >> 3, q = (tt & 7) * 16;
        async_b128(qB + r * 128 + q, qb,
                   (unsigned)((bh * SEQ + s0 + r) * DH) * 2 + q);
    }
    stage_kv(0, 0);

    float m_prev = -3.0e38f, l_run = 0.f;   // per lane: one query column
    f32x8 o0 = {}, o1 = {}, o2 = {}, o3 = {};   // O^T rows dh, col m_base+lo

    auto step = [&](const bf16_t (*ks_)[64], const bf16_t (*vs_)[64],
                    int t_next, unsigned boff_next, bool do_stage) {
        wait_async();
        __syncthreads();                    // K/V tile (and Q) visible
        if (do_stage) stage_kv(t_next, boff_next);   // prefetch overlaps WMMA

        // S^T = K Q^T : rows t (4 tiles), col m_base+lo.  K-dim = dh = 64.
        f32x8 sc[4] = {};
        #pragma unroll
        for (int kc = 0; kc < DH; kc += 32) {
            bf16x16 bq = frag_b(&q_s[m_base + lo][kc], hi);
            #pragma unroll
            for (int nt = 0; nt < 4; ++nt) {
                bf16x16 a = frag_a(&ks_[nt * 16 + lo][kc], hi);
                sc[nt] = wmma_bf16(a, bq, sc[nt]);
            }
        }

        // online softmax, fully in-register (hi-half partner lane holds the
        // other 32 t values -> one shfl_xor).
        float mx = m_prev;
        #pragma unroll
        for (int nt = 0; nt < 4; ++nt)
            for (int r = 0; r < 8; ++r) mx = fmaxf(mx, sc[nt][r]);
        mx = fmaxf(mx, __shfl_xor(mx, 16, 32));
        float alpha = __expf(m_prev - mx);
        float sum = 0.f;
        #pragma unroll
        for (int nt = 0; nt < 4; ++nt) {
            bf16x8 pv;
            for (int r = 0; r < 8; ++r) {
                float p = __expf(sc[nt][r] - mx);
                sum += p;
                pv[r] = (bf16_t)p;
            }
            // P^T stored as p_s[m][t]; this wave's rows only -> no barrier
            *(bf16x8*)&p_s[m_base + lo][nt * 16 + 8 * hi] = pv;
        }
        sum += __shfl_xor(sum, 16, 32);
        l_run = l_run * alpha + sum;
        m_prev = mx;

        // O^T = V^T P^T : rescale by alpha (one scalar/lane), then accumulate
        #pragma unroll
        for (int r = 0; r < 8; ++r) {
            o0[r] *= alpha; o1[r] *= alpha; o2[r] *= alpha; o3[r] *= alpha;
        }
        #pragma unroll
        for (int kc = 0; kc < 64; kc += 32) {
            bf16x16 bp = frag_b(&p_s[m_base + lo][kc], hi);
            o0 = wmma_bf16(frag_a(&vs_[lo][kc], hi),      bp, o0);
            o1 = wmma_bf16(frag_a(&vs_[16 + lo][kc], hi), bp, o1);
            o2 = wmma_bf16(frag_a(&vs_[32 + lo][kc], hi), bp, o2);
            o3 = wmma_bf16(frag_a(&vs_[48 + lo][kc], hi), bp, o3);
        }
        __syncthreads();                    // buf reads done before overwrite
    };

    for (int jt = 0; jt < SEQ / 64; jt += 2) {
        step(k_s[0], v_s[0], (jt + 1) * 64, 8192, jt + 1 < SEQ / 64);
        step(k_s[1], v_s[1], (jt + 2) * 64, 0,    jt + 2 < SEQ / 64);
    }

    // epilogue: O^T cols are query rows -> contiguous 16B stores per lane
    const float il = 1.0f / l_run;
    const int b = bh >> 4, h = bh & 15;
    const int s_g = s0 + m_base + lo;
    const size_t rowbase = ((size_t)b * SEQ + s_g) * EMB + (size_t)h * DH;
    f32x8 oo[4] = {o0, o1, o2, o3};
    #pragma unroll
    for (int mt = 0; mt < 4; ++mt) {
        bf16x8 ov;
        for (int r = 0; r < 8; ++r) ov[r] = (bf16_t)(oo[mt][r] * il);
        *(bf16x8*)&ob[rowbase + mt * 16 + 8 * hi] = ov;
    }
}

// ---------------------------------------------------------------------------
extern "C" void kernel_launch(void* const* d_in, const int* in_sizes, int n_in,
                              void* d_out, int out_size, void* d_ws,
                              size_t ws_size, hipStream_t stream) {
    const float* x     = (const float*)d_in[0];
    const float* w_qkv = (const float*)d_in[1];
    const float* w_out = (const float*)d_in[2];
    float* out = (float*)d_out;

    char* ws = (char*)d_ws;
    const size_t MB = 1u << 20;
    bf16_t* xh     = (bf16_t*)(ws);              // 16 MB  [8192][1024]
    bf16_t* wqkvT  = (bf16_t*)(ws + 16 * MB);    //  6 MB  [3072][1024]
    bf16_t* woutT  = (bf16_t*)(ws + 22 * MB);    //  2 MB  [1024][1024]
    bf16_t* qb     = (bf16_t*)(ws + 24 * MB);    // 16 MB  [bh][s][dh]
    bf16_t* kb     = (bf16_t*)(ws + 40 * MB);    // 16 MB  [bh][t][dh]
    bf16_t* vb     = (bf16_t*)(ws + 56 * MB);    // 16 MB  [bh][dh][t]
    bf16_t* ob     = (bf16_t*)(ws + 72 * MB);    // 16 MB  [b][s][1024]

    // Stage 0: one-time conversions
    cvt_kernel<<<(BATCH * SEQ * EMB) / (256 * 8), 256, 0, stream>>>(x, xh);
    cvt_tr_kernel<<<dim3(NQKV / 32, EMB / 32), 256, 0, stream>>>(w_qkv, wqkvT,
                                                                 EMB, NQKV);
    cvt_tr_kernel<<<dim3(EMB / 32, EMB / 32), 256, 0, stream>>>(w_out, woutT,
                                                                EMB, EMB);
    // Stage 1: QKV projection
    dim3 g1(NQKV / 128, (BATCH * SEQ) / 128);   // 24 x 64
    qkv_kernel<<<g1, 256, 0, stream>>>(xh, wqkvT, qb, kb, vb);
    // Stage 2: flash attention
    dim3 g2(SEQ / 128, BATCH * NUM_HEADS);      // 16 x 64
    attn_kernel<<<g2, 256, 0, stream>>>(qb, kb, vb, ob);
    // Stage 3: output projection
    dim3 g3(EMB / 128, (BATCH * SEQ) / 128);    //  8 x 64
    proj_kernel<<<g3, 256, 0, stream>>>(ob, woutT, out);
}
